// AttnBlock3D_37108517438324
// MI455X (gfx1250) — matrix operand order repaired and checked by reading, not compiled, not run on hardware
//
#include <hip/hip_runtime.h>
#include <hip/hip_bf16.h>
#include <stdint.h>

// ---------------- problem constants ----------------
#define C_DIM   512
#define T_DIM   8
#define HW      2304          // 48*48 tokens per frame
#define NT      18432         // T_DIM * HW
#define GROUPS  32
#define CPG     16            // channels per group
#define GRP_ELEMS (CPG * NT)  // 294912 contiguous floats per group

// ---------------- WMMA helpers ----------------
typedef __attribute__((ext_vector_type(16))) __bf16       v16bf;
typedef __attribute__((ext_vector_type(8)))  float        v8f;
typedef __attribute__((ext_vector_type(8)))  unsigned int v8u;

static __device__ __forceinline__ v8f wmma_bf16(v16bf a, v16bf b, v8f c) {
    return __builtin_amdgcn_wmma_f32_16x16x32_bf16(false, a, false, b,
                                                   (short)0, c, false, false);
}

static __device__ __forceinline__ unsigned short bf16_rn(float f) {
    union { float f; unsigned int u; } cv; cv.f = f;
    unsigned int u = cv.u;
    unsigned int r = u + 0x7FFFu + ((u >> 16) & 1u);
    return (unsigned short)(r >> 16);
}

static __device__ __forceinline__ v8f v8f_zero() {
    v8f z = {0.f, 0.f, 0.f, 0.f, 0.f, 0.f, 0.f, 0.f};
    return z;
}

// ---------------- async global -> LDS copy (CDNA5 path) ----------------
#if defined(__gfx1250__) && __has_builtin(__builtin_amdgcn_global_load_async_to_lds_b128)
#define HAVE_ASYNC_LDS 1
#endif

typedef int v4i128 __attribute__((vector_size(16)));
typedef __attribute__((address_space(1))) v4i128 as1_v4i;
typedef __attribute__((address_space(3))) v4i128 as3_v4i;

static __device__ __forceinline__ void cp16(void* lds, const void* g) {
#ifdef HAVE_ASYNC_LDS
    __builtin_amdgcn_global_load_async_to_lds_b128(
        (as1_v4i*)(g), (as3_v4i*)(lds), 0, 0);
#else
    *(uint4*)lds = *(const uint4*)g;
#endif
}
static __device__ __forceinline__ void cp_wait() {
#ifdef HAVE_ASYNC_LDS
#if __has_builtin(__builtin_amdgcn_s_wait_asynccnt)
    __builtin_amdgcn_s_wait_asynccnt(0);
#else
    asm volatile("s_wait_asynccnt 0" ::: "memory");
#endif
#endif
}

// ---------------- K0: weight fp32 -> bf16 ----------------
__global__ void cvt_w(const float* __restrict__ wq, const float* __restrict__ wk,
                      const float* __restrict__ wv, const float* __restrict__ wo,
                      unsigned short* oq, unsigned short* ok,
                      unsigned short* ov, unsigned short* oo) {
    int i = blockIdx.x * 256 + threadIdx.x;      // < C*C
    int which = blockIdx.y;
    const float* s = (which == 0) ? wq : (which == 1) ? wk : (which == 2) ? wv : wo;
    unsigned short* d = (which == 0) ? oq : (which == 1) ? ok : (which == 2) ? ov : oo;
    d[i] = bf16_rn(s[i]);
}

// ---------------- K1: groupnorm statistics ----------------
__global__ void gn_stats(const float* __restrict__ x, float* __restrict__ stats) {
    int g = blockIdx.x;                           // 32 groups, contiguous chunks
    const float* p = x + (size_t)g * GRP_ELEMS;
    float s = 0.f, sq = 0.f;
    for (int i = threadIdx.x; i < GRP_ELEMS; i += 256) {
        float v = p[i]; s += v; sq += v * v;
    }
    __shared__ float ls[256], lq[256];
    ls[threadIdx.x] = s; lq[threadIdx.x] = sq;
    __syncthreads();
    for (int st = 128; st > 0; st >>= 1) {
        if (threadIdx.x < st) {
            ls[threadIdx.x] += ls[threadIdx.x + st];
            lq[threadIdx.x] += lq[threadIdx.x + st];
        }
        __syncthreads();
    }
    if (threadIdx.x == 0) {
        float inv = 1.0f / (float)GRP_ELEMS;
        float mean = ls[0] * inv;
        float var  = lq[0] * inv - mean * mean;
        stats[2 * g]     = mean;
        stats[2 * g + 1] = rsqrtf(var + 1e-6f);
    }
}

// ---------------- K2: normalize + transpose to token-major bf16 ----------------
__global__ void gn_apply(const float* __restrict__ x, const float* __restrict__ gamma,
                         const float* __restrict__ beta, const float* __restrict__ stats,
                         unsigned short* __restrict__ hn) {
    int cBase   = blockIdx.x * 16;    // 32 blocks == one group each
    int tokBase = blockIdx.y * 64;    // 288 blocks
    int g = blockIdx.x;
    float mu = stats[2 * g], rs = stats[2 * g + 1];
    __shared__ float tile[16][65];
    int tid = threadIdx.x;
    int cl = tid >> 4;                // 0..15
    int t4 = (tid & 15) * 4;          // 0..60
    const float* xp = x + (size_t)(cBase + cl) * NT + tokBase + t4;
    float gm = gamma[cBase + cl], bt = beta[cBase + cl];
    #pragma unroll
    for (int i = 0; i < 4; i++) tile[cl][t4 + i] = (xp[i] - mu) * rs * gm + bt;
    __syncthreads();
    int tl = tid >> 2;                // 0..63
    int c4 = (tid & 3) * 4;
    unsigned short* hp = hn + (size_t)(tokBase + tl) * C_DIM + cBase + c4;
    #pragma unroll
    for (int i = 0; i < 4; i++) hp[i] = bf16_rn(tile[c4 + i][tl]);
}

// ---------------- K3: fused Q/K/V projection GEMM ----------------
// block tile: 64 tokens x 64 outputs x 3 targets; 8 waves, 6 WMMA tiles per wave.
__global__ __launch_bounds__(256) void qkv_gemm(
    const unsigned short* __restrict__ hn,
    const unsigned short* __restrict__ wqb, const unsigned short* __restrict__ wkb,
    const unsigned short* __restrict__ wvb,
    const float* __restrict__ bq, const float* __restrict__ bk, const float* __restrict__ bv,
    unsigned short* __restrict__ qo, unsigned short* __restrict__ ko,
    unsigned short* __restrict__ vo) {
    __shared__ __align__(16) unsigned short A[64][32];
    __shared__ __align__(16) unsigned short Bt[3][64][32];
    int tokBase = blockIdx.x * 64;
    int oBase   = blockIdx.y * 64;
    int tid = threadIdx.x;
    int wave = tid >> 5, lane = tid & 31;
    int half = lane >> 4, l16 = lane & 15;
    int mi = wave & 3, nh = wave >> 2;       // nh in {0,1}, cols nh*32..

    // hoisted per-thread staging addresses (1 A-chunk + 3 B-chunks per k-step)
    int sr = tid >> 2, sc8 = (tid & 3) * 8;
    const unsigned short* gA  = hn  + (size_t)(tokBase + sr) * C_DIM + sc8;
    const unsigned short* gB0 = wqb + (size_t)(oBase + sr) * C_DIM + sc8;
    const unsigned short* gB1 = wkb + (size_t)(oBase + sr) * C_DIM + sc8;
    const unsigned short* gB2 = wvb + (size_t)(oBase + sr) * C_DIM + sc8;

    v8f acc[3][2];
    #pragma unroll
    for (int tg = 0; tg < 3; tg++)
        #pragma unroll
        for (int u = 0; u < 2; u++) acc[tg][u] = v8f_zero();

    for (int kc = 0; kc < C_DIM; kc += 32) {
        cp16(&A[sr][sc8],     gA  + kc);
        cp16(&Bt[0][sr][sc8], gB0 + kc);
        cp16(&Bt[1][sr][sc8], gB1 + kc);
        cp16(&Bt[2][sr][sc8], gB2 + kc);
        cp_wait();
        __syncthreads();
        v8u au;
        #pragma unroll
        for (int v = 0; v < 8; v++) {
            int kk = (half << 4) + (v << 1);
            au[v] = *(const unsigned int*)&A[mi * 16 + l16][kk];
        }
        v16bf a = __builtin_bit_cast(v16bf, au);
        #pragma unroll
        for (int tg = 0; tg < 3; tg++) {
            #pragma unroll
            for (int u = 0; u < 2; u++) {
                int n = (nh * 2 + u) * 16 + l16;
                v8u bu;
                #pragma unroll
                for (int v = 0; v < 8; v++) {
                    int kk = (half << 4) + (v << 1);
                    bu[v] = *(const unsigned int*)&Bt[tg][n][kk];
                }
                v16bf b = __builtin_bit_cast(v16bf, bu);
                acc[tg][u] = wmma_bf16(a, b, acc[tg][u]);
            }
        }
        __syncthreads();
    }
    // epilogue: bias + bf16 store
    #pragma unroll
    for (int tg = 0; tg < 3; tg++) {
        const float* bp = (tg == 0) ? bq : (tg == 1) ? bk : bv;
        unsigned short* dp = (tg == 0) ? qo : (tg == 1) ? ko : vo;
        #pragma unroll
        for (int u = 0; u < 2; u++) {
            int o = oBase + (nh * 2 + u) * 16 + l16;
            float bs = bp[o];
            #pragma unroll
            for (int r = 0; r < 8; r++) {
                int tok = tokBase + mi * 16 + (half << 3) + r;
                dp[(size_t)tok * C_DIM + o] = bf16_rn(acc[tg][u][r] + bs);
            }
        }
    }
}

// ---------------- K4: per-frame flash attention ----------------
static constexpr unsigned SM_Q = 0;                          // 64x512 bf16 = 65536
static constexpr unsigned SM_K = 65536;                      // 64x512 bf16 = 65536
static constexpr unsigned SM_V = 131072;                     // 512x66 ushort = 67584
static constexpr unsigned SM_S = 198656;                     // 64x64 f32 = 16384
static constexpr unsigned SM_P = 215040;                     // 64x64 bf16 = 8192
static constexpr unsigned SM_R = 223232;                     // 3*64 f32 = 768
static constexpr unsigned SM_TOTAL = 224000;

__global__ __launch_bounds__(256) void attn_flash(
    const unsigned short* __restrict__ qg, const unsigned short* __restrict__ kg,
    const unsigned short* __restrict__ vg, unsigned short* __restrict__ ao) {
    extern __shared__ __align__(16) char smem[];
    unsigned short (*Q)[C_DIM] = (unsigned short(*)[C_DIM])(smem + SM_Q);
    unsigned short (*K)[C_DIM] = (unsigned short(*)[C_DIM])(smem + SM_K);
    unsigned short (*V)[66]    = (unsigned short(*)[66])(smem + SM_V);   // [c][key]
    float (*S)[64]             = (float(*)[64])(smem + SM_S);
    unsigned short (*P)[64]    = (unsigned short(*)[64])(smem + SM_P);
    float* mrow = (float*)(smem + SM_R);
    float* lrow = mrow + 64;
    float* arow = lrow + 64;

    int t = blockIdx.x;
    int tokQ = t * HW + blockIdx.y * 64;
    int tid = threadIdx.x;
    int wave = tid >> 5, lane = tid & 31;
    int half = lane >> 4, l16 = lane & 15;
    int miS  = wave >> 1, niS0 = (wave & 1) * 2;   // S phase: 2 tiles / wave
    int miO  = wave >> 1, cHalf = wave & 1;        // O phase: 16 col tiles / wave

    // staging coordinates (hoisted): each thread copies 16 chunks of 16B
    int rg  = tid >> 6;              // 0..3
    int cc8 = (tid & 63) * 8;        // 0..504

    // stage Q block once (async)
    #pragma unroll 4
    for (int p = 0; p < 16; p++) {
        int r = 4 * p + rg;
        cp16(&Q[r][cc8], &qg[(size_t)(tokQ + r) * C_DIM + cc8]);
    }
    if (tid < 64) { mrow[tid] = -1e30f; lrow[tid] = 0.f; }

    v8f accO[16];
    #pragma unroll
    for (int i = 0; i < 16; i++) accO[i] = v8f_zero();

    const float scale = 0.044194173824159216f;   // 512^-0.5

    for (int j = 0; j < HW / 64; j++) {
        int tokK = t * HW + j * 64;
        // stage K (async) then V transposed (sync scatter) - overlapped
        #pragma unroll 4
        for (int p = 0; p < 16; p++) {
            int r = 4 * p + rg;
            cp16(&K[r][cc8], &kg[(size_t)(tokK + r) * C_DIM + cc8]);
        }
        #pragma unroll 4
        for (int p = 0; p < 16; p++) {
            int r = 4 * p + rg;
            uint4 vv = *(const uint4*)&vg[(size_t)(tokK + r) * C_DIM + cc8];
            const unsigned short* vs = (const unsigned short*)&vv;
            #pragma unroll
            for (int e = 0; e < 8; e++) V[cc8 + e][r] = vs[e];
        }
        cp_wait();
        __syncthreads();

        // ---- S = scale * Q K^T (each wave: 2 of 16 tiles) ----
        v8f s0 = v8f_zero(), s1 = v8f_zero();
        for (int kc = 0; kc < C_DIM; kc += 32) {
            v8u au;
            #pragma unroll
            for (int v = 0; v < 8; v++) {
                int kk = kc + (half << 4) + (v << 1);
                au[v] = *(const unsigned int*)&Q[miS * 16 + l16][kk];
            }
            v16bf a = __builtin_bit_cast(v16bf, au);
            #pragma unroll
            for (int u = 0; u < 2; u++) {
                int keyCol = (niS0 + u) * 16 + l16;
                v8u bu;
                #pragma unroll
                for (int v = 0; v < 8; v++) {
                    int kk = kc + (half << 4) + (v << 1);
                    bu[v] = *(const unsigned int*)&K[keyCol][kk];
                }
                v16bf b = __builtin_bit_cast(v16bf, bu);
                if (u == 0) s0 = wmma_bf16(a, b, s0);
                else        s1 = wmma_bf16(a, b, s1);
            }
        }
        #pragma unroll
        for (int r = 0; r < 8; r++) {
            int m = miS * 16 + (half << 3) + r;
            S[m][niS0 * 16 + l16]       = s0[r] * scale;
            S[m][(niS0 + 1) * 16 + l16] = s1[r] * scale;
        }
        __syncthreads();

        // ---- online softmax: 4 lanes per row, 16 cols each ----
        {
            int r = tid >> 2, q4 = (tid & 3) << 4;
            float vals[16];
            float mx = -1e30f;
            #pragma unroll
            for (int i = 0; i < 16; i++) { vals[i] = S[r][q4 + i]; mx = fmaxf(mx, vals[i]); }
            mx = fmaxf(mx, __shfl_xor(mx, 1));
            mx = fmaxf(mx, __shfl_xor(mx, 2));
            float mold = mrow[r];
            float mnew = fmaxf(mold, mx);
            float sum = 0.f;
            #pragma unroll
            for (int i = 0; i < 16; i++) {
                float p = __expf(vals[i] - mnew);
                P[r][q4 + i] = bf16_rn(p);
                sum += p;
            }
            sum += __shfl_xor(sum, 1);
            sum += __shfl_xor(sum, 2);
            if ((tid & 3) == 0) {
                float alpha = __expf(mold - mnew);
                mrow[r] = mnew;
                lrow[r] = lrow[r] * alpha + sum;
                arow[r] = alpha;
            }
        }
        __syncthreads();

        // ---- rescale O and accumulate O += P @ V ----
        float alph[8];
        #pragma unroll
        for (int r = 0; r < 8; r++) alph[r] = arow[miO * 16 + (half << 3) + r];
        #pragma unroll
        for (int nt = 0; nt < 16; nt++)
            #pragma unroll
            for (int r = 0; r < 8; r++) accO[nt][r] *= alph[r];

        v16bf pa[2];
        #pragma unroll
        for (int ks = 0; ks < 2; ks++) {
            v8u au;
            #pragma unroll
            for (int v = 0; v < 8; v++) {
                int kk = ks * 32 + (half << 4) + (v << 1);
                au[v] = *(const unsigned int*)&P[miO * 16 + l16][kk];
            }
            pa[ks] = __builtin_bit_cast(v16bf, au);
        }
        #pragma unroll
        for (int nt = 0; nt < 16; nt++) {
            int c = cHalf * 256 + nt * 16 + l16;
            #pragma unroll
            for (int ks = 0; ks < 2; ks++) {
                v8u bu;
                #pragma unroll
                for (int v = 0; v < 8; v++) {
                    int kk = ks * 32 + (half << 4) + (v << 1);
                    bu[v] = *(const unsigned int*)&V[c][kk];
                }
                v16bf b = __builtin_bit_cast(v16bf, bu);
                accO[nt] = wmma_bf16(pa[ks], b, accO[nt]);
            }
        }
        __syncthreads();
    }

    // ---- finalize: divide by l, store bf16 token-major ----
    float linv[8];
    #pragma unroll
    for (int r = 0; r < 8; r++) linv[r] = 1.f / lrow[miO * 16 + (half << 3) + r];
    #pragma unroll
    for (int nt = 0; nt < 16; nt++) {
        int c = cHalf * 256 + nt * 16 + l16;
        #pragma unroll
        for (int r = 0; r < 8; r++) {
            int tok = tokQ + miO * 16 + (half << 3) + r;
            ao[(size_t)tok * C_DIM + c] = bf16_rn(accO[nt][r] * linv[r]);
        }
    }
}

// ---------------- K5: output projection + bias + residual ----------------
__global__ __launch_bounds__(256) void out_proj(
    const unsigned short* __restrict__ aob, const unsigned short* __restrict__ wob,
    const float* __restrict__ bo, const float* __restrict__ x,
    float* __restrict__ out) {
    __shared__ __align__(16) unsigned short A[64][32];
    __shared__ __align__(16) unsigned short Bt[64][32];
    __shared__ float Ot[64][65];   // [tok][o]
    int tokBase = blockIdx.x * 64;
    int oBase   = blockIdx.y * 64;
    int tid = threadIdx.x;
    int wave = tid >> 5, lane = tid & 31;
    int half = lane >> 4, l16 = lane & 15;
    int mi = wave >> 1, ni0 = (wave & 1) * 2;

    int sr = tid >> 2, sc8 = (tid & 3) * 8;
    const unsigned short* gA = aob + (size_t)(tokBase + sr) * C_DIM + sc8;
    const unsigned short* gB = wob + (size_t)(oBase + sr) * C_DIM + sc8;

    v8f acc0 = v8f_zero(), acc1 = v8f_zero();
    for (int kc = 0; kc < C_DIM; kc += 32) {
        cp16(&A[sr][sc8],  gA + kc);
        cp16(&Bt[sr][sc8], gB + kc);
        cp_wait();
        __syncthreads();
        v8u au;
        #pragma unroll
        for (int v = 0; v < 8; v++) {
            int kk = (half << 4) + (v << 1);
            au[v] = *(const unsigned int*)&A[mi * 16 + l16][kk];
        }
        v16bf a = __builtin_bit_cast(v16bf, au);
        #pragma unroll
        for (int u = 0; u < 2; u++) {
            v8u bu;
            #pragma unroll
            for (int v = 0; v < 8; v++) {
                int kk = (half << 4) + (v << 1);
                bu[v] = *(const unsigned int*)&Bt[(ni0 + u) * 16 + l16][kk];
            }
            v16bf b = __builtin_bit_cast(v16bf, bu);
            if (u == 0) acc0 = wmma_bf16(a, b, acc0);
            else        acc1 = wmma_bf16(a, b, acc1);
        }
        __syncthreads();
    }
    #pragma unroll
    for (int r = 0; r < 8; r++) {
        int m = mi * 16 + (half << 3) + r;
        Ot[m][ni0 * 16 + l16]       = acc0[r];
        Ot[m][(ni0 + 1) * 16 + l16] = acc1[r];
    }
    __syncthreads();
    int oL = tid >> 2;
    int tL = (tid & 3) * 16;
    int o = oBase + oL;
    float bs = bo[o];
    const float* xp = x + (size_t)o * NT + tokBase + tL;
    float* op = out + (size_t)o * NT + tokBase + tL;
    #pragma unroll
    for (int i = 0; i < 16; i++) op[i] = xp[i] + Ot[tL + i][oL] + bs;
}

// ---------------- workspace layout ----------------
static constexpr size_t SZ_TOKC  = (size_t)NT * C_DIM * 2;   // 18,874,368 B
static constexpr size_t SZ_W     = (size_t)C_DIM * C_DIM * 2;
static constexpr size_t OFF_STATS = 0;
static constexpr size_t OFF_HN = 256;
static constexpr size_t OFF_WQ = OFF_HN + SZ_TOKC;
static constexpr size_t OFF_WK = OFF_WQ + SZ_W;
static constexpr size_t OFF_WV = OFF_WK + SZ_W;
static constexpr size_t OFF_WO = OFF_WV + SZ_W;
static constexpr size_t OFF_Q  = OFF_WO + SZ_W;
static constexpr size_t OFF_K  = OFF_Q + SZ_TOKC;
static constexpr size_t OFF_V  = OFF_K + SZ_TOKC;
static constexpr size_t OFF_AO = OFF_V + SZ_TOKC;

extern "C" void kernel_launch(void* const* d_in, const int* in_sizes, int n_in,
                              void* d_out, int out_size, void* d_ws, size_t ws_size,
                              hipStream_t stream) {
    (void)in_sizes; (void)n_in; (void)out_size; (void)ws_size;
    const float* x     = (const float*)d_in[0];
    const float* gamma = (const float*)d_in[1];
    const float* beta  = (const float*)d_in[2];
    const float* wq    = (const float*)d_in[3];
    const float* bq    = (const float*)d_in[4];
    const float* wk    = (const float*)d_in[5];
    const float* bk    = (const float*)d_in[6];
    const float* wv    = (const float*)d_in[7];
    const float* bv    = (const float*)d_in[8];
    const float* wo    = (const float*)d_in[9];
    const float* bo    = (const float*)d_in[10];
    float* out = (float*)d_out;

    char* ws = (char*)d_ws;
    float* stats = (float*)(ws + OFF_STATS);
    unsigned short* hn  = (unsigned short*)(ws + OFF_HN);
    unsigned short* wqb = (unsigned short*)(ws + OFF_WQ);
    unsigned short* wkb = (unsigned short*)(ws + OFF_WK);
    unsigned short* wvb = (unsigned short*)(ws + OFF_WV);
    unsigned short* wob = (unsigned short*)(ws + OFF_WO);
    unsigned short* qb  = (unsigned short*)(ws + OFF_Q);
    unsigned short* kb  = (unsigned short*)(ws + OFF_K);
    unsigned short* vb  = (unsigned short*)(ws + OFF_V);
    unsigned short* aob = (unsigned short*)(ws + OFF_AO);

    cvt_w<<<dim3(C_DIM * C_DIM / 256, 4), 256, 0, stream>>>(wq, wk, wv, wo,
                                                            wqb, wkb, wvb, wob);
    gn_stats<<<GROUPS, 256, 0, stream>>>(x, stats);
    gn_apply<<<dim3(C_DIM / 16, NT / 64), 256, 0, stream>>>(x, gamma, beta, stats, hn);
    qkv_gemm<<<dim3(NT / 64, C_DIM / 64), 256, 0, stream>>>(hn, wqb, wkb, wvb,
                                                            bq, bk, bv, qb, kb, vb);
    attn_flash<<<dim3(T_DIM, HW / 64), 256, SM_TOTAL, stream>>>(qb, kb, vb, aob);
    out_proj<<<dim3(NT / 64, C_DIM / 64), 256, 0, stream>>>(aob, wob, bo, x, out);
}
